// LeadingZeroDetector28_23407571764185
// MI455X (gfx1250) — matrix operand order repaired and checked by reading, not compiled, not run on hardware
//
#include <hip/hip_runtime.h>

// 28-bit leading-zero detector, 2^21 rows.
// Streaming (memory-bound, ~277 MB total): float4 NT loads -> bitmask ->
// p = clz(mask)-4 (p==28 for all-zero rows) -> one-hot(p)[16x32 f16] x
// posmask[32x16 f16] via v_wmma_f32_16x16x32_f16 -> 5 output bits per row.

typedef float     __attribute__((ext_vector_type(4)))  f4;
typedef _Float16  __attribute__((ext_vector_type(16))) v16h;
typedef float     __attribute__((ext_vector_type(8)))  v8f;

union HU { unsigned int u[8]; v16h h; };

#define BITS 28
#define LZCW 5

__global__ __launch_bounds__(256) void lzd28_wmma_kernel(const float* __restrict__ X,
                                                         float* __restrict__ out,
                                                         int nRows) {
    const int lane    = threadIdx.x & 31;
    const int wave    = (blockIdx.x * blockDim.x + threadIdx.x) >> 5;
    const int rowBase = wave * 32;
    if (rowBase >= nRows) return;   // wave-uniform exit: EXEC stays all-ones below

    // ---- B matrix (32x16 f16): B[k][n] = bit n (MSB-first) of k for k<=28, n<5.
    // Register layout: VGPR j, lanes 0-15 -> N=lane,   K=2j,2j+1
    //                          lanes16-31 -> N=lane-16, K=16+2j,16+2j+1
    const int n  = lane & 15;
    const int kb = (lane >> 4) << 4;
    HU B;
#pragma unroll
    for (int j = 0; j < 8; ++j) {
        int k0 = kb + 2 * j;
        unsigned lo = (n < LZCW && k0     <= BITS) ? ((unsigned)(k0     >> (4 - n)) & 1u) : 0u;
        unsigned hi = (n < LZCW && k0 + 1 <= BITS) ? ((unsigned)((k0+1) >> (4 - n)) & 1u) : 0u;
        B.u[j] = lo * 0x3C00u | hi * 0x3C000000u;   // packed f16 {lo, hi}, 1.0h = 0x3C00
    }

    // ---- Load this lane's row (112 B, 16B-aligned since 28*4=112=7*16), build mask.
    int row = rowBase + lane;
    if (row >= nRows) row = nRows - 1;              // clamp keeps EXEC full on ragged tail
    const f4* rp = (const f4*)(X + (size_t)row * BITS);
    unsigned msk = 0u;
#pragma unroll
    for (int q = 0; q < 7; ++q) {
        f4 v = __builtin_nontemporal_load(rp + q);  // streaming: NT hint, no reuse
        msk = (msk << 1) | (v.x != 0.0f ? 1u : 0u);
        msk = (msk << 1) | (v.y != 0.0f ? 1u : 0u);
        msk = (msk << 1) | (v.z != 0.0f ? 1u : 0u);
        msk = (msk << 1) | (v.w != 0.0f ? 1u : 0u);
    }
    // x_0 sits at bit 27; first-one index = clz-4; clz(0)=32 -> p=28 (all-zero code).
    int p = __clz((int)msk) - 4;

    // Both half-waves need row M's position (A stripes K across the two halves).
    const int m  = lane & 15;
    const int p0 = __shfl(p, m,      32);   // rows rowBase+0..15
    const int p1 = __shfl(p, m + 16, 32);   // rows rowBase+16..31

    // ---- A matrices (16x32 f16 one-hot rows).
    // Lane L (half=L>>4, M=L&15): VGPR j -> K = 16*half+2j, 16*half+2j+1
    const int half8 = (lane >> 4) << 3;
    HU A0, A1;
#pragma unroll
    for (int j = 0; j < 8; ++j) {
        unsigned k0 = (unsigned)(kb + 2 * j);   // always even
        A0.u[j] = ((unsigned)(p0 & ~1) == k0) ? (0x3C00u << ((p0 & 1) << 4)) : 0u;
        A1.u[j] = ((unsigned)(p1 & ~1) == k0) ? (0x3C00u << ((p1 & 1) << 4)) : 0u;
    }

    // ---- D = A x B (+0): exact since operands are {0,1} and exactly one K hits.
    v8f c0 = {};
    v8f d0 = __builtin_amdgcn_wmma_f32_16x16x32_f16(false, A0.h, false, B.h,
                                                    (short)0, c0, false, false);
    v8f d1 = __builtin_amdgcn_wmma_f32_16x16x32_f16(false, A1.h, false, B.h,
                                                    (short)0, c0, false, false);

    // ---- Store. D layout: VGPR v, lane L -> element (M = v + 8*(L>>4), N = L&15).
    // Common case (nRows is a multiple of 32): branch-free unguarded stores under a
    // single wave-uniform check; guarded path only for a generic ragged tail tile.
    const int  mOff     = (lane >> 4) << 3;
    const bool fullTile = (rowBase + 32 <= nRows);  // wave-uniform -> scalar branch
    if (fullTile) {
        if (n < LZCW) {                              // one exec-mask set, then 16 stores
            float* o0 = out + (size_t)(rowBase + mOff) * LZCW + n;
#pragma unroll
            for (int v = 0; v < 8; ++v) {
                __builtin_nontemporal_store(d0[v], o0 + (size_t)v * LZCW);
                __builtin_nontemporal_store(d1[v], o0 + (size_t)(16 + v) * LZCW);
            }
        }
    } else {
        if (n < LZCW) {
#pragma unroll
            for (int v = 0; v < 8; ++v) {
                int r0 = rowBase + mOff + v;
                int r1 = r0 + 16;
                if (r0 < nRows) __builtin_nontemporal_store(d0[v], &out[(size_t)r0 * LZCW + n]);
                if (r1 < nRows) __builtin_nontemporal_store(d1[v], &out[(size_t)r1 * LZCW + n]);
            }
        }
    }
}

extern "C" void kernel_launch(void* const* d_in, const int* in_sizes, int n_in,
                              void* d_out, int out_size, void* d_ws, size_t ws_size,
                              hipStream_t stream) {
    const float* X   = (const float*)d_in[0];
    float*       out = (float*)d_out;
    const int nRows  = in_sizes[0] / BITS;          // 2,097,152

    // One wave per 32-row tile; 256-thread blocks = 8 waves/block (wave32).
    const int tiles         = (nRows + 31) / 32;
    const int threads       = 256;
    const int wavesPerBlock = threads / 32;
    const int blocks        = (tiles + wavesPerBlock - 1) / wavesPerBlock;

    lzd28_wmma_kernel<<<blocks, threads, 0, stream>>>(X, out, nRows);
}
